// GaussianRenderer_58677843198015
// MI455X (gfx1250) — matrix operand order repaired and checked by reading, not compiled, not run on hardware
//
#include <hip/hip_runtime.h>
#include <hip/hip_bf16.h>

typedef __attribute__((ext_vector_type(16))) _Float16 v16h;
typedef __attribute__((ext_vector_type(8)))  float    v8f;
typedef __attribute__((ext_vector_type(4)))  float    f32x4;

#define GS_H 64
#define GS_W 64
#define GS_N 512
#define GS_B 32

#ifndef LOG2E_F
#define LOG2E_F 1.4426950408889634f
#endif

// grid = B*4 blocks, block = 128 threads (4 waves).
// Each block: batch b = blockIdx.x>>2, image-quarter q = blockIdx.x&3 (16 rows).
// Each wave: 16 pixel-groups of 16 pixels; per group a WMMA chain over N=512 (K=32/step).
__global__ __launch_bounds__(128)
void GaussianRenderer_58677843198015_kernel(const float* __restrict__ data,     // (B,N,8)
                                            const float* __restrict__ opacity,  // (N,1)
                                            float* __restrict__ out)            // (B,3,H,W)
{
    __shared__ float    prm[GS_N * 8];     // {a2, b, c2, px, py, pad,pad,pad} per n  (16 KB)
    __shared__ _Float16 pkC[16][GS_N];     // color*opacity per WMMA "channel" row    (16 KB)

    const int t    = threadIdx.x;
    const int lane = t & 31;
    const int wv   = t >> 5;
    const int b    = blockIdx.x >> 2;
    const int q    = blockIdx.x & 3;

    // ---- phase 0: zero the color rows (channels 3..15 stay zero) ----
    {
        uint4* z = (uint4*)&pkC[0][0];
        // 16*512 halves = 16 KB = 1024 uint4
        for (int i = t; i < 1024; i += 128) z[i] = make_uint4(0u, 0u, 0u, 0u);
    }
    __syncthreads();

    // ---- phase 1: per-gaussian parameter setup (shared across whole block) ----
    for (int n = t; n < GS_N; n += 128) {
        const float* d = data + ((size_t)b * GS_N + n) * 8;
        f32x4 d0 = *(const f32x4*)d;        // x, y, sc0, sc1
        f32x4 d1 = *(const f32x4*)(d + 4);  // theta_raw, r, g, bcol
        float px = 0.5f * ((tanhf(d0.x) + 1.0f) * (float)GS_W - 1.0f);
        float py = 0.5f * ((tanhf(d0.y) + 1.0f) * (float)GS_H - 1.0f);
        float s0 = fabsf(d0.z) + 0.3f;
        float s1 = fabsf(d0.w) + 0.3f;
        float s0q = s0 * s0, s1q = s1 * s1;
        // theta = sigmoid(d4) * 2*pi; v_sin/v_cos take input in revolutions,
        // so the sigmoid value feeds the HW trig directly.
        float sg = __builtin_amdgcn_rcpf(1.0f + __builtin_amdgcn_exp2f(-LOG2E_F * d1.x));
        float c  = __builtin_amdgcn_cosf(sg);
        float s  = __builtin_amdgcn_sinf(sg);
        float cov_a = c * c * s0q + s * s * s1q;
        float cov_b = c * s * (s0q - s1q);
        float cov_c = s * s * s0q + c * c * s1q;
        float rdet  = __builtin_amdgcn_rcpf(cov_a * cov_c - cov_b * cov_b);
        float* P = &prm[n * 8];
        P[0] = 0.5f * cov_c * rdet;   // a2 = 0.5*con_a
        P[1] = -cov_b * rdet;         // b  = con_b
        P[2] = 0.5f * cov_a * rdet;   // c2 = 0.5*con_c
        P[3] = px;
        P[4] = py;
        float op = opacity[n];        // fold opacity into color (linear in the einsum)
        pkC[0][n] = (_Float16)(d1.y * op);
        pkC[1][n] = (_Float16)(d1.z * op);
        pkC[2][n] = (_Float16)(d1.w * op);
    }
    __syncthreads();

    const int m  = lane & 15;         // A/D row: pixel index within 16-pixel group
    const int hi = lane >> 4;         // lane-half
    const int ch = m;                 // D column: channel for this lane
    const int kA = hi ? 16 : 0;       // A-matrix K base for this half (contiguous k = 16h + s)
    const int kB = hi ? 16 : 0;       // B-matrix K base for this half

    for (int gi = 0; gi < 16; ++gi) {
        const int   g   = wv * 16 + gi;
        const int   row = q * 16 + (g >> 2);
        const int   cb  = (g & 3) * 16;
        const float X   = (float)(cb + m);
        const float Y   = (float)row;

        v8f acc = {};
        for (int kc = 0; kc < GS_N; kc += 32) {
            // B fragment: 16 consecutive f16 of this channel's color row (2x ds_load_b128)
            v16h bm = *(const v16h*)&pkC[ch][kc + kB];
            // A fragment: compute 16 alphas directly in the WMMA A layout
            v16h am;
#pragma unroll
            for (int v = 0; v < 8; ++v) {
                const int kbase = kc + kA + (v << 1);
#pragma unroll
                for (int j = 0; j < 2; ++j) {
                    const float* P  = &prm[(kbase + j) * 8];
                    f32x4 p4 = *(const f32x4*)P;     // a2, b, c2, px
                    float py = P[4];
                    float dx = X - p4.w;
                    float dy = Y - py;
                    float sgm = (p4.x * dx + p4.y * dy) * dx + p4.z * dy * dy;
                    am[2 * v + j] = (_Float16)__builtin_amdgcn_exp2f(-LOG2E_F * sgm);
                }
            }
            // D(16 px x 16 ch) += A(16 px x 32 n) * B(32 n x 16 ch)
            acc = __builtin_amdgcn_wmma_f32_16x16x32_f16(
                false, am, false, bm, (short)0, acc, false, false);
        }

        // D layout: VGPR r holds (M=r, lanes 0-15) / (M=r+8, lanes 16-31) -> 8 contiguous w
        if (ch < 3) {
            float* op = out + (((size_t)b * 3 + ch) * GS_H + row) * GS_W + cb + hi * 8;
            f32x4 lo = { acc[0], acc[1], acc[2], acc[3] };
            f32x4 hv = { acc[4], acc[5], acc[6], acc[7] };
            *(f32x4*)op       = lo;
            *(f32x4*)(op + 4) = hv;
        }
    }
}

extern "C" void kernel_launch(void* const* d_in, const int* in_sizes, int n_in,
                              void* d_out, int out_size, void* d_ws, size_t ws_size,
                              hipStream_t stream) {
    (void)in_sizes; (void)n_in; (void)d_ws; (void)ws_size; (void)out_size;
    const float* data    = (const float*)d_in[0];
    const float* opacity = (const float*)d_in[1];
    float*       out     = (float*)d_out;
    dim3 grid(GS_B * 4);
    dim3 block(128);
    hipLaunchKernelGGL(GaussianRenderer_58677843198015_kernel, grid, block, 0, stream,
                       data, opacity, out);
}
